// PertAggregator_9869834846789
// MI455X (gfx1250) — matrix operand order repaired and checked by reading, not compiled, not run on hardware
//
#include <hip/hip_runtime.h>

// CDNA5 / gfx1250 fused MLP (Linear->ReLU->Linear) + sorted segment-sum.
// bf16 V_WMMA_F32_16X16X32_BF16 with f32 accumulation; weights pre-packed to
// WMMA B-fragment layout in d_ws (needs 512 KB of workspace).

typedef __attribute__((ext_vector_type(16))) __bf16 v16bf;
typedef __attribute__((ext_vector_type(8)))  __bf16 v8bf;
typedef __attribute__((ext_vector_type(8)))  float  v8f;

#define EMB 256
#define HID 512
#define NOUT 256
#define ROWS_PER_BLK 64
#define XS_STRIDE 264   // bf16/row (256 + 8 pad) -> 528 B, 16B-aligned rows
#define HS_STRIDE 520   // bf16/row (512 + 8 pad) -> 1040 B, 16B-aligned rows
#define YS_STRIDE 260   // f32/row  (256 + 4 pad) to avoid bank conflicts
#define XS_BYTES (ROWS_PER_BLK * XS_STRIDE * 2)
#define HS_BYTES (ROWS_PER_BLK * HS_STRIDE * 2)
#define YS_BYTES (ROWS_PER_BLK * YS_STRIDE * 4)
#define SMEM_BYTES (XS_BYTES + HS_BYTES + YS_BYTES + ROWS_PER_BLK * 4)

// Pack fp32 weight W[K][Ncols] into bf16 WMMA B-fragments:
// dst[((kt*NT + nt)*32 + lane)*16 + e] = bf16(W[kt*32 + 8*(lane>=16) + (e&7) + 16*(e>>3)][nt*16 + lane%16])
__global__ __launch_bounds__(256)
void pack_w_kernel(const float* __restrict__ W, __bf16* __restrict__ dst,
                   int K, int Ncols) {
    int tid = blockIdx.x * 256 + threadIdx.x;
    if (tid >= K * Ncols) return;
    int NT   = Ncols >> 4;
    int e    = tid & 15;
    int lane = (tid >> 4) & 31;
    int rest = tid >> 9;               // kt*NT + nt
    int nt   = rest % NT;
    int kt   = rest / NT;
    int k = kt * 32 + ((lane >> 4) << 3) + (e & 7) + ((e >> 3) << 4);
    int n = nt * 16 + (lane & 15);
    dst[tid] = (__bf16)W[(size_t)k * Ncols + n];
}

__global__ __launch_bounds__(256)
void zero_out_kernel(float* __restrict__ out, int n) {
    int i = blockIdx.x * 256 + threadIdx.x;
    if (i < n) out[i] = 0.0f;
}

__global__ __launch_bounds__(256)
void PertAggregator_9869834846789_kernel(const float* __restrict__ X,
                                         const float* __restrict__ b1,
                                         const float* __restrict__ b2,
                                         const __bf16* __restrict__ W1p,
                                         const __bf16* __restrict__ W2p,
                                         const int* __restrict__ seg,
                                         float* __restrict__ out) {
    extern __shared__ char smem[];
    __bf16* Xs = (__bf16*)smem;
    __bf16* Hs = (__bf16*)(smem + XS_BYTES);
    float*  Ys = (float*)(smem + XS_BYTES + HS_BYTES);
    int*    Ss = (int*)(smem + XS_BYTES + HS_BYTES + YS_BYTES);

    const int tid  = threadIdx.x;
    const int lane = tid & 31;
    const int wave = tid >> 5;
    const size_t rowBase = (size_t)blockIdx.x * ROWS_PER_BLK;

    // ---- Stage X tile (64 x 256 fp32) into LDS as bf16, padded rows ----
    for (int i = tid; i < ROWS_PER_BLK * (EMB / 2); i += 256) {
        int r  = i >> 7;        // EMB/2 = 128 float2 per row
        int c2 = i & 127;
        float2 v = ((const float2*)(X + (rowBase + r) * EMB))[c2];
        union { unsigned u; __bf16 h[2]; } p;
        p.h[0] = (__bf16)v.x;
        p.h[1] = (__bf16)v.y;
        ((unsigned*)Xs)[r * (XS_STRIDE / 2) + c2] = p.u;
    }
    if (tid < ROWS_PER_BLK) Ss[tid] = seg[rowBase + tid];
    __syncthreads();

    const int nsub = lane & 15;
    const int kbA  = (lane >> 4) << 3;  // A-fragment K-offset: 0 or 8
    const int mHi  = (lane >> 4) << 3;  // C/D row offset: 0 or 8

    // ---- GEMM1: H = relu(X @ W1 + b1), bf16 H into LDS ----
    for (int nn = 0; nn < 4; ++nn) {
        const int nt = wave * 4 + nn;   // 32 HID column tiles / 8 waves
        v16bf bfr[8];
#pragma unroll
        for (int kt = 0; kt < 8; ++kt)
            bfr[kt] = *(const v16bf*)(W1p +
                        (size_t)((kt * 32 + nt) * 32 + lane) * 16);
        const int n = nt * 16 + nsub;
        const float bias = b1[n];
        for (int strip = 0; strip < 4; ++strip) {
            const int m = strip * 16 + nsub;
            v8f acc = {};
#pragma unroll
            for (int kt = 0; kt < 8; ++kt) {
                const int k0 = kt * 32 + kbA;
                v8bf lo = *(const v8bf*)(Xs + m * XS_STRIDE + k0);
                v8bf hi = *(const v8bf*)(Xs + m * XS_STRIDE + k0 + 16);
                v16bf a;
#pragma unroll
                for (int i = 0; i < 8; ++i) { a[i] = lo[i]; a[8 + i] = hi[i]; }
                acc = __builtin_amdgcn_wmma_f32_16x16x32_bf16(
                        false, a, false, bfr[kt], (short)0, acc, false, false);
            }
            const int mr = strip * 16 + mHi;
#pragma unroll
            for (int v = 0; v < 8; ++v) {
                float h = acc[v] + bias;
                h = h > 0.0f ? h : 0.0f;
                Hs[(mr + v) * HS_STRIDE + n] = (__bf16)h;
            }
        }
    }
    __syncthreads();

    // ---- GEMM2: Y = H @ W2 + b2, f32 Y into LDS ----
    for (int nn = 0; nn < 2; ++nn) {
        const int nt = wave * 2 + nn;   // 16 OUT column tiles / 8 waves
        v16bf bfr[16];
#pragma unroll
        for (int kt = 0; kt < 16; ++kt)
            bfr[kt] = *(const v16bf*)(W2p +
                        (size_t)((kt * 16 + nt) * 32 + lane) * 16);
        const int n = nt * 16 + nsub;
        const float bias = b2[n];
        for (int strip = 0; strip < 4; ++strip) {
            const int m = strip * 16 + nsub;
            v8f acc = {};
#pragma unroll
            for (int kt = 0; kt < 16; ++kt) {
                const int k0 = kt * 32 + kbA;
                v8bf lo = *(const v8bf*)(Hs + m * HS_STRIDE + k0);
                v8bf hi = *(const v8bf*)(Hs + m * HS_STRIDE + k0 + 16);
                v16bf a;
#pragma unroll
                for (int i = 0; i < 8; ++i) { a[i] = lo[i]; a[8 + i] = hi[i]; }
                acc = __builtin_amdgcn_wmma_f32_16x16x32_bf16(
                        false, a, false, bfr[kt], (short)0, acc, false, false);
            }
            const int mr = strip * 16 + mHi;
#pragma unroll
            for (int v = 0; v < 8; ++v)
                Ys[(mr + v) * YS_STRIDE + n] = acc[v] + bias;
        }
    }
    __syncthreads();

    // ---- Sorted segment-sum inside block, one atomic per boundary ----
    const int col = tid;                // 256 threads == NOUT columns
    float acc = 0.0f;
    int cur = Ss[0];
    for (int r = 0; r < ROWS_PER_BLK; ++r) {
        int s = Ss[r];
        if (s != cur) {
            atomicAdd(out + (size_t)cur * NOUT + col, acc);
            acc = 0.0f;
            cur = s;
        }
        acc += Ys[r * YS_STRIDE + col];
    }
    atomicAdd(out + (size_t)cur * NOUT + col, acc);
}

extern "C" void kernel_launch(void* const* d_in, const int* in_sizes, int n_in,
                              void* d_out, int out_size, void* d_ws, size_t ws_size,
                              hipStream_t stream) {
    const float* X  = (const float*)d_in[0];
    const float* W1 = (const float*)d_in[1];
    const float* b1 = (const float*)d_in[2];
    const float* W2 = (const float*)d_in[3];
    const float* b2 = (const float*)d_in[4];
    const int*  seg = (const int*)d_in[5];
    float* out = (float*)d_out;

    const int N = in_sizes[0] / EMB;          // 400000
    __bf16* W1p = (__bf16*)d_ws;                                   // 256 KB
    __bf16* W2p = (__bf16*)((char*)d_ws + (size_t)EMB * HID * 2);  // 256 KB

    // Pack weights to bf16 WMMA fragment layout (hot in L2 afterwards).
    pack_w_kernel<<<(EMB * HID + 255) / 256, 256, 0, stream>>>(W1, W1p, EMB, HID);
    pack_w_kernel<<<(HID * NOUT + 255) / 256, 256, 0, stream>>>(W2, W2p, HID, NOUT);

    // d_out is poisoned by the harness; zero before atomic accumulation.
    zero_out_kernel<<<(out_size + 255) / 256, 256, 0, stream>>>(out, out_size);

    const int nblk = N / ROWS_PER_BLK;        // 6250 (N divisible by 64)
    PertAggregator_9869834846789_kernel<<<nblk, 256, SMEM_BYTES, stream>>>(
        X, b1, b2, W1p, W2p, seg, out);
}
